// CausalAttention_72756745994824
// MI455X (gfx1250) — matrix operand order repaired and checked by reading, not compiled, not run on hardware
//
#include <hip/hip_runtime.h>

// ---------------------------------------------------------------------------
// CDNA5 (gfx1250) implementation: bf16 WMMA 16x16x32 everywhere, f32 accum.
// Edge-MLP (dominant GEMM) stages weights in LDS via async global->LDS copies.
// ---------------------------------------------------------------------------

typedef __bf16 bf16;
typedef __attribute__((ext_vector_type(8)))  bf16  bf16x8;
typedef __attribute__((ext_vector_type(16))) bf16  bf16x16;
typedef __attribute__((ext_vector_type(8)))  float f32x8;

__device__ __forceinline__ f32x8 zero8() {
  f32x8 z = {0.f, 0.f, 0.f, 0.f, 0.f, 0.f, 0.f, 0.f};
  return z;
}

// A-matrix 16x32 bf16 fragment (ISA 7.12.2): lane L holds row M=L%16.
// half = L>=16: K halves 0..7 -> K = half*8 + i ; halves 8..15 -> K = half*8+16+i
__device__ __forceinline__ bf16x16 frag_from_row(const bf16* rowp) {
  const int half = (threadIdx.x >> 4) & 1;
  const bf16* p = rowp + half * 8;
  bf16x8 lo = *(const bf16x8*)(p);
  bf16x8 hi = *(const bf16x8*)(p + 16);
  bf16x16 r;
#pragma unroll
  for (int i = 0; i < 8; ++i) { r[i] = lo[i]; r[i + 8] = hi[i]; }
  return r;
}

__device__ __forceinline__ bf16x16 load_frag_a(const bf16* base, int ld) {
  return frag_from_row(base + (size_t)(threadIdx.x & 15) * ld);
}

// B-matrix 32x16 bf16 fragment, from BT stored row-major [N][K]:
// lane L: column n = L%16, kg = L>=16: halves 0..7 -> K = kg*8 + i ; halves 8..15 -> K = kg*8+16+i
__device__ __forceinline__ bf16x16 load_frag_b(const bf16* base, int ld) {
  const int n  = threadIdx.x & 15;
  const int kg = (threadIdx.x >> 4) & 1;
  const bf16* p = base + (size_t)n * ld + kg * 8;
  bf16x8 lo = *(const bf16x8*)(p);
  bf16x8 hi = *(const bf16x8*)(p + 16);
  bf16x16 r;
#pragma unroll
  for (int i = 0; i < 8; ++i) { r[i] = lo[i]; r[i + 8] = hi[i]; }
  return r;
}

__device__ __forceinline__ f32x8 wmma32(bf16x16 a, bf16x16 b, f32x8 c) {
  // 8 args: (neg_a, A, neg_b, B, c_mod, C, reuse_a, reuse_b)
  return __builtin_amdgcn_wmma_f32_16x16x32_bf16(false, a, false, b, (short)0, c,
                                                 false, false);
}

__device__ __forceinline__ float sigmoidf_(float x) {
  return 1.0f / (1.0f + __expf(-x));
}

// gfx1250 async global->LDS 16B copy (ASYNCcnt-tracked, no VGPR round trip).
// lds_off: wave-relative LDS byte address (low 32 bits of a generic shared ptr,
// per ISA 10.2 aperture truncation). gaddr: 64-bit global byte address.
__device__ __forceinline__ void async_g2l_b128(unsigned lds_off,
                                               unsigned long long gaddr) {
  asm volatile("global_load_async_to_lds_b128 %0, %1, off"
               :: "v"(lds_off), "v"(gaddr)
               : "memory");
}

__device__ __forceinline__ void wait_asynccnt0() {
  asm volatile("s_wait_asynccnt 0" ::: "memory");
}

// ---------------------------------------------------------------------------
// Elementwise f32 -> bf16 convert
// ---------------------------------------------------------------------------
__global__ void k_cvt(const float* __restrict__ in, bf16* __restrict__ out, int n) {
  int i = blockIdx.x * blockDim.x + threadIdx.x;
  if (i < n) out[i] = (bf16)in[i];
}

// Transposed convert: in [K,N] f32 row-major -> out [N,K] bf16 row-major
__global__ void k_cvtT(const float* __restrict__ in, bf16* __restrict__ out,
                       int K, int N) {
  int i = blockIdx.x * blockDim.x + threadIdx.x;
  if (i < K * N) {
    int k = i / N, n = i % N;
    out[(size_t)n * K + k] = (bf16)in[i];
  }
}

// ---------------------------------------------------------------------------
// Edge-gather MLP: v[b,e] = sigmoid( relu([f[src]|f[dst]] @ w1 + b1) . w2 + b2 )
// scattered to mask[b, src[e], dst[e]].
// Block: 64 edges x N=256, K=512. 8 waves = 2(M) x 4(N); wave: 2 mtiles x 4 ntiles.
// Per k-step the 256x32 w1T panel (16 KB) is staged in LDS with async copies and
// shared by all waves; B fragments then come from LDS (ds_load_b128).
// ---------------------------------------------------------------------------
__global__ __launch_bounds__(256)
void k_edge_mlp(const bf16* __restrict__ feat, const int* __restrict__ src,
                const int* __restrict__ dst, const bf16* __restrict__ W1T,
                const float* __restrict__ b1, const float* __restrict__ w2,
                const float* __restrict__ b2, float* __restrict__ maskout) {
  __shared__ __align__(16) bf16 Bt[256 * 32];   // [n][k] panel, 16 KB
  __shared__ float part[8][64];
  const int lane = threadIdx.x & 31;
  const int wv = threadIdx.x >> 5;
  const int wm = wv >> 2;       // 0..1 : 32-row half
  const int wn = wv & 3;        // 0..3 : 64-wide N slice
  const int b = blockIdx.y;
  const int e0 = blockIdx.x * 64;

  // Per-lane gathered feature rows for this wave's two mtiles.
  const bf16* ps[2];
  const bf16* pd[2];
#pragma unroll
  for (int mt = 0; mt < 2; ++mt) {
    const int e = e0 + wm * 32 + mt * 16 + (lane & 15);
    ps[mt] = feat + ((size_t)b * 512 + src[e]) * 256;
    pd[mt] = feat + ((size_t)b * 512 + dst[e]) * 256;
  }

  const unsigned lds_base = (unsigned)(unsigned long long)(&Bt[0]);

  f32x8 acc[2][4];
#pragma unroll
  for (int mt = 0; mt < 2; ++mt)
#pragma unroll
    for (int nt = 0; nt < 4; ++nt) acc[mt][nt] = zero8();

  for (int kk = 0; kk < 512; kk += 32) {
    // Stage W1T[0:256][kk:kk+32] -> LDS via async copies (4 x b128 per thread).
#pragma unroll
    for (int i = 0; i < 4; ++i) {
      const int l = threadIdx.x + i * 256;      // 0..1023 16B chunks
      const int n = l >> 2;
      const int ch = l & 3;
      async_g2l_b128(lds_base + (unsigned)(n * 64 + ch * 16),
                     (unsigned long long)(W1T + (size_t)n * 512 + kk + ch * 8));
    }
    wait_asynccnt0();
    __syncthreads();   // panel visible to all waves

    bf16x16 af[2];
#pragma unroll
    for (int mt = 0; mt < 2; ++mt) {
      const bf16* arow = (kk < 256) ? (ps[mt] + kk) : (pd[mt] + (kk - 256));
      af[mt] = frag_from_row(arow);
      __builtin_prefetch(arow + 32, 0, 1);
    }
#pragma unroll
    for (int nt = 0; nt < 4; ++nt) {
      bf16x16 bfb = load_frag_b(Bt + (wn * 64 + nt * 16) * 32, 32);
#pragma unroll
      for (int mt = 0; mt < 2; ++mt)
        acc[mt][nt] = wmma32(af[mt], bfb, acc[mt][nt]);
    }
    __syncthreads();   // all reads done before next panel overwrites
  }

  // relu + bias + dot with w2 over this wave's 64 columns
  float partial[2][8];
#pragma unroll
  for (int mt = 0; mt < 2; ++mt)
#pragma unroll
    for (int g = 0; g < 8; ++g) partial[mt][g] = 0.f;
#pragma unroll
  for (int nt = 0; nt < 4; ++nt) {
    const int c = wn * 64 + nt * 16 + (lane & 15);
    const float w2v = w2[c];
    const float b1v = b1[c];
#pragma unroll
    for (int mt = 0; mt < 2; ++mt)
#pragma unroll
      for (int g = 0; g < 8; ++g) {
        float hv = acc[mt][nt][g] + b1v;
        hv = fmaxf(hv, 0.f);
        partial[mt][g] += hv * w2v;
      }
  }
  // reduce across the 16 lanes of each half-wave (same M rows)
#pragma unroll
  for (int mk = 1; mk < 16; mk <<= 1)
#pragma unroll
    for (int mt = 0; mt < 2; ++mt)
#pragma unroll
      for (int g = 0; g < 8; ++g)
        partial[mt][g] += __shfl_xor(partial[mt][g], mk, 32);

  if ((lane & 15) == 0) {
    const int rb = (lane >> 4) << 3;
#pragma unroll
    for (int mt = 0; mt < 2; ++mt)
#pragma unroll
      for (int g = 0; g < 8; ++g)
        part[wv][wm * 32 + mt * 16 + rb + g] = partial[mt][g];
  }
  __syncthreads();
  if (threadIdx.x < 64) {
    const int r = threadIdx.x;
    const int wmr = r >> 5;
    float s = part[wmr * 4 + 0][r] + part[wmr * 4 + 1][r] +
              part[wmr * 4 + 2][r] + part[wmr * 4 + 3][r];
    const float v = sigmoidf_(s + b2[0]);
    const int e2 = e0 + r;
    maskout[(size_t)b * 262144 + (size_t)src[e2] * 512 + dst[e2]] = v;
  }
}

// ---------------------------------------------------------------------------
// QKV projection: [8192,256] x in_w^T[768,256] + in_b, scatter epilogue:
//   Q -> [b,h,i,d], K -> [b,h,j,d], V -> transposed [b,h,d,j]
// Columns 0..255 use Aq (query source); 256..767 use Akv (key/value source).
// ---------------------------------------------------------------------------
__global__ __launch_bounds__(256)
void k_qkv(const bf16* __restrict__ Aq, const bf16* __restrict__ Akv,
           const bf16* __restrict__ BT, const float* __restrict__ bias,
           bf16* __restrict__ qb, bf16* __restrict__ kb, bf16* __restrict__ vtb) {
  const int lane = threadIdx.x & 31;
  const int wv = threadIdx.x >> 5;
  const int wm = wv >> 2, wn = wv & 3;
  const int nblock = blockIdx.y;                       // 0..5
  const bf16* A = (nblock < 2) ? Aq : Akv;
  const int rowbase = blockIdx.x * 64 + wm * 32;
  const int colbase = nblock * 128 + wn * 32;

  f32x8 acc[2][2];
#pragma unroll
  for (int mt = 0; mt < 2; ++mt)
#pragma unroll
    for (int nt = 0; nt < 2; ++nt) acc[mt][nt] = zero8();

  for (int kk = 0; kk < 256; kk += 32) {
    bf16x16 af[2], bfb[2];
#pragma unroll
    for (int mt = 0; mt < 2; ++mt)
      af[mt] = load_frag_a(A + (size_t)(rowbase + mt * 16) * 256 + kk, 256);
#pragma unroll
    for (int nt = 0; nt < 2; ++nt)
      bfb[nt] = load_frag_b(BT + (size_t)(colbase + nt * 16) * 256 + kk, 256);
#pragma unroll
    for (int mt = 0; mt < 2; ++mt)
#pragma unroll
      for (int nt = 0; nt < 2; ++nt)
        acc[mt][nt] = wmma32(af[mt], bfb[nt], acc[mt][nt]);
  }

  const int Mg = (lane >> 4) * 8;
  const int Nl = lane & 15;
#pragma unroll
  for (int mt = 0; mt < 2; ++mt)
#pragma unroll
    for (int nt = 0; nt < 2; ++nt)
#pragma unroll
      for (int g = 0; g < 8; ++g) {
        const int r = rowbase + mt * 16 + Mg + g;
        const int c = colbase + nt * 16 + Nl;
        const float v = acc[mt][nt][g] + bias[c];
        const int bb = r >> 9, i = r & 511;
        const bf16 bv = (bf16)v;
        if (c < 256) {
          qb[(((size_t)(bb * 8 + (c >> 5))) * 512 + i) * 32 + (c & 31)] = bv;
        } else if (c < 512) {
          const int cc = c - 256;
          kb[(((size_t)(bb * 8 + (cc >> 5))) * 512 + i) * 32 + (cc & 31)] = bv;
        } else {
          const int cc = c - 512;
          vtb[(((size_t)(bb * 8 + (cc >> 5))) * 32 + (cc & 31)) * 512 + i] = bv;
        }
      }
}

// ---------------------------------------------------------------------------
// Fused attention per (b, h, 16-query tile per wave):
//   S = Q K^T * scale + mask  (one WMMA per 16-key tile; DH=32 = one K-step)
//   row softmax over all 512 keys in LDS, probs -> bf16 LDS
//   O = P V  (Vt serves directly as WMMA B operand)
// Block = 64 threads (2 waves). LDS: 2*(16*512*4) f32 + 2*(16*512*2) bf16.
// ---------------------------------------------------------------------------
__global__ __launch_bounds__(64)
void k_attn(const bf16* __restrict__ q, const bf16* __restrict__ k,
            const bf16* __restrict__ vt, const float* __restrict__ mask,
            bf16* __restrict__ obuf) {
  extern __shared__ char smem[];
  const int lane = threadIdx.x & 31;
  const int wave = threadIdx.x >> 5;
  const int b = blockIdx.z, h = blockIdx.y;
  const int qbase = (blockIdx.x * 2 + wave) * 16;
  float* srow = (float*)(smem + (size_t)wave * 32768);              // 16 x 512 f32
  bf16* prow = (bf16*)(smem + 65536 + (size_t)wave * 16384);        // 16 x 512 bf16
  const size_t bh = (size_t)b * 8 + h;
  const bf16x16 aq = load_frag_a(q + (bh * 512 + qbase) * 32, 32);
  const int Mg = (lane >> 4) * 8;
  const int Nl = lane & 15;
  const float scale = 0.17677669529663687f;   // 1/sqrt(32)

  for (int nt = 0; nt < 32; ++nt) {
    bf16x16 bk = load_frag_b(k + (bh * 512 + nt * 16) * 32, 32);
    f32x8 acc = wmma32(aq, bk, zero8());
#pragma unroll
    for (int g = 0; g < 8; ++g) {
      float s = acc[g] * scale;
      if (mask)
        s += mask[((size_t)b * 512 + (qbase + Mg + g)) * 512 + nt * 16 + Nl];
      srow[(Mg + g) * 512 + nt * 16 + Nl] = s;
    }
  }
  __syncthreads();

  for (int r = 0; r < 16; ++r) {
    float v[16];
#pragma unroll
    for (int i = 0; i < 16; ++i) v[i] = srow[r * 512 + lane * 16 + i];
    float mx = v[0];
#pragma unroll
    for (int i = 1; i < 16; ++i) mx = fmaxf(mx, v[i]);
#pragma unroll
    for (int mk = 1; mk < 32; mk <<= 1) mx = fmaxf(mx, __shfl_xor(mx, mk, 32));
    float sum = 0.f;
#pragma unroll
    for (int i = 0; i < 16; ++i) { v[i] = __expf(v[i] - mx); sum += v[i]; }
#pragma unroll
    for (int mk = 1; mk < 32; mk <<= 1) sum += __shfl_xor(sum, mk, 32);
    const float inv = 1.0f / sum;
#pragma unroll
    for (int i = 0; i < 16; ++i) prow[r * 512 + lane * 16 + i] = (bf16)(v[i] * inv);
  }
  __syncthreads();

#pragma unroll
  for (int dt = 0; dt < 2; ++dt) {
    f32x8 acc = zero8();
    for (int kt = 0; kt < 32; ++kt) {
      bf16x16 ap = load_frag_a(prow + kt * 32, 512);
      bf16x16 bv = load_frag_b(vt + (bh * 32 + dt * 16) * 512 + kt * 32, 512);
      acc = wmma32(ap, bv, acc);
    }
#pragma unroll
    for (int g = 0; g < 8; ++g) {
      const int row = b * 512 + qbase + Mg + g;
      const int col = h * 32 + dt * 16 + Nl;
      obuf[(size_t)row * 256 + col] = (bf16)acc[g];
    }
  }
}

// ---------------------------------------------------------------------------
// Out-projection GEMM: [8192,256] x out_w(BT)[256,256] + out_b.
// Writes f32 to d_out and (optionally) a bf16 copy for downstream kernels.
// ---------------------------------------------------------------------------
__global__ __launch_bounds__(256)
void k_gemm_dual(const bf16* __restrict__ A, const bf16* __restrict__ BT,
                 const float* __restrict__ bias, float* __restrict__ outf,
                 bf16* __restrict__ outb) {
  const int lane = threadIdx.x & 31;
  const int wv = threadIdx.x >> 5;
  const int wm = wv >> 2, wn = wv & 3;
  const int rowbase = blockIdx.x * 64 + wm * 32;
  const int colbase = blockIdx.y * 128 + wn * 32;

  f32x8 acc[2][2];
#pragma unroll
  for (int mt = 0; mt < 2; ++mt)
#pragma unroll
    for (int nt = 0; nt < 2; ++nt) acc[mt][nt] = zero8();

  for (int kk = 0; kk < 256; kk += 32) {
    bf16x16 af[2], bfb[2];
#pragma unroll
    for (int mt = 0; mt < 2; ++mt)
      af[mt] = load_frag_a(A + (size_t)(rowbase + mt * 16) * 256 + kk, 256);
#pragma unroll
    for (int nt = 0; nt < 2; ++nt)
      bfb[nt] = load_frag_b(BT + (size_t)(colbase + nt * 16) * 256 + kk, 256);
#pragma unroll
    for (int mt = 0; mt < 2; ++mt)
#pragma unroll
      for (int nt = 0; nt < 2; ++nt)
        acc[mt][nt] = wmma32(af[mt], bfb[nt], acc[mt][nt]);
  }

  const int Mg = (lane >> 4) * 8;
  const int Nl = lane & 15;
#pragma unroll
  for (int mt = 0; mt < 2; ++mt)
#pragma unroll
    for (int nt = 0; nt < 2; ++nt)
#pragma unroll
      for (int g = 0; g < 8; ++g) {
        const int r = rowbase + mt * 16 + Mg + g;
        const int c = colbase + nt * 16 + Nl;
        const float v = acc[mt][nt][g] + bias[c];
        outf[(size_t)r * 256 + c] = v;
        if (outb) outb[(size_t)r * 256 + c] = (bf16)v;
      }
}

// ---------------------------------------------------------------------------
// Consistency head: out[r] = sigmoid( relu(X[r,:] @ cw1 + cb1) . cw2 + cb2 )
// Block: 32 rows, N=256, K=256.
// ---------------------------------------------------------------------------
__global__ __launch_bounds__(256)
void k_row_mlp(const bf16* __restrict__ X, const bf16* __restrict__ W1T,
               const float* __restrict__ b1, const float* __restrict__ w2,
               const float* __restrict__ b2, float* __restrict__ out) {
  __shared__ float part[8][32];
  const int lane = threadIdx.x & 31;
  const int wv = threadIdx.x >> 5;
  const int wm = wv >> 2, wn = wv & 3;
  const int r0 = blockIdx.x * 32;
  const int r = r0 + wm * 16 + (lane & 15);
  const bf16* rp = X + (size_t)r * 256;

  f32x8 acc[4];
#pragma unroll
  for (int nt = 0; nt < 4; ++nt) acc[nt] = zero8();

  for (int kk = 0; kk < 256; kk += 32) {
    bf16x16 af = frag_from_row(rp + kk);
#pragma unroll
    for (int nt = 0; nt < 4; ++nt) {
      bf16x16 bfb = load_frag_b(W1T + (size_t)(wn * 64 + nt * 16) * 256 + kk, 256);
      acc[nt] = wmma32(af, bfb, acc[nt]);
    }
  }

  float partial[8];
#pragma unroll
  for (int g = 0; g < 8; ++g) partial[g] = 0.f;
#pragma unroll
  for (int nt = 0; nt < 4; ++nt) {
    const int c = wn * 64 + nt * 16 + (lane & 15);
    const float w2v = w2[c];
    const float b1v = b1[c];
#pragma unroll
    for (int g = 0; g < 8; ++g) {
      float hv = acc[nt][g] + b1v;
      hv = fmaxf(hv, 0.f);
      partial[g] += hv * w2v;
    }
  }
#pragma unroll
  for (int mk = 1; mk < 16; mk <<= 1)
#pragma unroll
    for (int g = 0; g < 8; ++g) partial[g] += __shfl_xor(partial[g], mk, 32);

  if ((lane & 15) == 0) {
    const int rb = wm * 16 + ((lane >> 4) << 3);
#pragma unroll
    for (int g = 0; g < 8; ++g) part[wv][rb + g] = partial[g];
  }
  __syncthreads();
  if (threadIdx.x < 32) {
    const int rr = threadIdx.x;
    const int wmr = rr >> 4;
    float s = part[wmr * 4 + 0][rr] + part[wmr * 4 + 1][rr] +
              part[wmr * 4 + 2][rr] + part[wmr * 4 + 3][rr];
    out[r0 + rr] = sigmoidf_(s + b2[0]);
  }
}

// ---------------------------------------------------------------------------
// Host orchestration
// ---------------------------------------------------------------------------
extern "C" void kernel_launch(void* const* d_in, const int* in_sizes, int n_in,
                              void* d_out, int out_size, void* d_ws, size_t ws_size,
                              hipStream_t stream) {
  (void)in_sizes; (void)n_in; (void)out_size; (void)ws_size;
  const float* img   = (const float*)d_in[0];
  const float* txt   = (const float*)d_in[1];
  const int*   src   = (const int*)d_in[2];
  const int*   dst   = (const int*)d_in[3];
  const float* w1    = (const float*)d_in[4];
  const float* b1    = (const float*)d_in[5];
  const float* w2    = (const float*)d_in[6];
  const float* b2    = (const float*)d_in[7];
  const float* in_w  = (const float*)d_in[8];
  const float* in_b  = (const float*)d_in[9];
  const float* out_w = (const float*)d_in[10];
  const float* out_b = (const float*)d_in[11];
  const float* cw1   = (const float*)d_in[12];
  const float* cb1   = (const float*)d_in[13];
  const float* cw2   = (const float*)d_in[14];
  const float* cb2   = (const float*)d_in[15];

  float* outf = (float*)d_out;
  float* att_img = outf + 0;
  float* att_txt = outf + 2097152;
  float* cross   = outf + 4194304;
  float* wimg    = outf + 6291456;
  float* wtxt    = outf + 10485760;
  float* cimg    = outf + 14680064;
  float* ctxt    = outf + 14688256;

  char* ws = (char*)d_ws;
  bf16* fimg  = (bf16*)(ws + 0);          // 4 MB  [8192,256]
  bf16* ftxt  = (bf16*)(ws + 4194304);    // 4 MB
  bf16* inwb  = (bf16*)(ws + 8388608);    // 384 KB  [768,256] (already BT layout)
  bf16* outwb = (bf16*)(ws + 8781824);    // 128 KB  [256,256] (already BT layout)
  bf16* w1T   = (bf16*)(ws + 8912896);    // 256 KB  [256,512]
  bf16* cw1T  = (bf16*)(ws + 9175040);    // 128 KB  [256,256]
  bf16* qb    = (bf16*)(ws + 9306112);    // 4 MB  [b,h,i,d]
  bf16* kb    = (bf16*)(ws + 13500416);   // 4 MB  [b,h,j,d]
  bf16* vtb   = (bf16*)(ws + 17694720);   // 4 MB  [b,h,d,j]
  bf16* ob    = (bf16*)(ws + 21889024);   // 4 MB  pre-outproj O
  bf16* aimgb = (bf16*)(ws + 26083328);   // 4 MB  attended_img bf16
  bf16* atxtb = (bf16*)(ws + 30277632);   // 4 MB  attended_text bf16

  // Masks are scatter-written: zero them first.
  hipMemsetAsync(wimg, 0, (size_t)2 * 16 * 512 * 512 * sizeof(float), stream);

  // Precision conversions / weight transposes.
  k_cvt<<<(2097152 + 255) / 256, 256, 0, stream>>>(img, fimg, 2097152);
  k_cvt<<<(2097152 + 255) / 256, 256, 0, stream>>>(txt, ftxt, 2097152);
  k_cvt<<<(196608 + 255) / 256, 256, 0, stream>>>(in_w, inwb, 196608);
  k_cvt<<<(65536 + 255) / 256, 256, 0, stream>>>(out_w, outwb, 65536);
  k_cvtT<<<(512 * 256 + 255) / 256, 256, 0, stream>>>(w1, w1T, 512, 256);
  k_cvtT<<<(256 * 256 + 255) / 256, 256, 0, stream>>>(cw1, cw1T, 256, 256);

  // Causal weight masks (dominant GEMM: 2 x [262144,512]x[512,256]).
  k_edge_mlp<<<dim3(256, 16), 256, 0, stream>>>(fimg, src, dst, w1T, b1, w2, b2, wimg);
  k_edge_mlp<<<dim3(256, 16), 256, 0, stream>>>(ftxt, src, dst, w1T, b1, w2, b2, wtxt);

  const size_t attn_lds = 98304;  // 2 waves * (32KB f32 scores + 16KB bf16 probs)

  // Self-attention (img), masked.
  k_qkv<<<dim3(128, 6), 256, 0, stream>>>(fimg, fimg, inwb, in_b, qb, kb, vtb);
  k_attn<<<dim3(16, 8, 16), 64, attn_lds, stream>>>(qb, kb, vtb, wimg, ob);
  k_gemm_dual<<<dim3(128, 2), 256, 0, stream>>>(ob, outwb, out_b, att_img, aimgb);

  // Self-attention (text), masked.
  k_qkv<<<dim3(128, 6), 256, 0, stream>>>(ftxt, ftxt, inwb, in_b, qb, kb, vtb);
  k_attn<<<dim3(16, 8, 16), 64, attn_lds, stream>>>(qb, kb, vtb, wtxt, ob);
  k_gemm_dual<<<dim3(128, 2), 256, 0, stream>>>(ob, outwb, out_b, att_txt, atxtb);

  // Consistency heads.
  k_row_mlp<<<256, 256, 0, stream>>>(aimgb, cw1T, cb1, cw2, cb2, cimg);
  k_row_mlp<<<256, 256, 0, stream>>>(atxtb, cw1T, cb1, cw2, cb2, ctxt);

  // Cross-modal attention: Q from attended_img, K/V from attended_text, no mask.
  k_qkv<<<dim3(128, 6), 256, 0, stream>>>(aimgb, atxtb, inwb, in_b, qb, kb, vtb);
  k_attn<<<dim3(16, 8, 16), 64, attn_lds, stream>>>(qb, kb, vtb, nullptr, ob);
  k_gemm_dual<<<dim3(128, 2), 256, 0, stream>>>(ob, outwb, out_b, cross, nullptr);
}
